// CrossModal_11587821764885
// MI455X (gfx1250) — matrix operand order repaired and checked by reading, not compiled, not run on hardware
//
#include <hip/hip_runtime.h>

// ---------------------------------------------------------------------------
// Problem constants (from reference)
// ---------------------------------------------------------------------------
constexpr int B_   = 8;
constexpr int SQ_  = 1024;
constexpr int SK_  = 1024;
constexpr int D_   = 512;
constexpr int H_   = 8;
constexpr int HD_  = 64;     // D_/H_
constexpr int DFF_ = 2048;
constexpr int L_   = 2;

typedef __attribute__((ext_vector_type(16))) _Float16 v16h;
typedef __attribute__((ext_vector_type(8)))  float    v8f;
typedef __attribute__((ext_vector_type(4)))  unsigned int v4u;
typedef int v4i __attribute__((__vector_size__(4 * sizeof(int))));

union F16Frag {          // 32 bytes = 16 halves = one WMMA f16 A/B fragment
    v4u  u[2];
    v16h h;
};

// ---------------------------------------------------------------------------
// CDNA5 async global->LDS path (ASYNCcnt-tracked), with safe fallback.
// Builtin signature (from probe diagnostic): (v4i* src, v4i* lds, Ii, Ii)
// ---------------------------------------------------------------------------
#if defined(__AMDGCN__)
#  if __has_builtin(__builtin_amdgcn_global_load_async_to_lds_b128)
#    define HAVE_ASYNC_LDS 1
#  endif
#endif
#ifndef HAVE_ASYNC_LDS
#  define HAVE_ASYNC_LDS 0
#endif

// copy 16 bytes global -> LDS
__device__ __forceinline__ void cp16_g2l(const _Float16* g, _Float16* l) {
#if HAVE_ASYNC_LDS
    __builtin_amdgcn_global_load_async_to_lds_b128(
        (v4i*)const_cast<_Float16*>(g), (v4i*)l, 0, 0);
#else
    *(v4u*)l = *(const v4u*)g;
#endif
}

__device__ __forceinline__ void wait_async_all() {
#if HAVE_ASYNC_LDS
#  if __has_builtin(__builtin_amdgcn_s_wait_asynccnt)
    __builtin_amdgcn_s_wait_asynccnt(0);
#  else
    asm volatile("s_wait_asynccnt 0" ::: "memory");
#  endif
#endif
}

__device__ __forceinline__ float gelu_exact(float x) {
    return 0.5f * x * (1.0f + erff(x * 0.70710678118654752f));
}

// ---------------------------------------------------------------------------
// NT WMMA GEMM:  C[M,N] = A[M,K] * BT[N,K]^T (+bias) (+epilogue)
//   A   : f16 row-major [M][K]      (lda)
//   BT  : f16 row-major [N][K]      (ldb)  == B stored transposed
//   block = 256 threads = 8 wave32s; block tile 128x128; wave tile 32x64
//   Double-buffered LDS, tiles staged via GLOBAL_LOAD_ASYNC_TO_LDS_B128.
//   EPI: 0 = f32 (+bias); 2 = f16 (+bias); 3 = f16 (+bias, exact gelu);
//        4 = f16 transposed store for V^T:  out[(b*D + col)*SK + s]
//   grid.z batching with split strides: off = (z/zdiv)*s0 + (z%zdiv)*s1
//   Requires M % 128 == 0, K % 32 == 0, N % 16 == 0.
// ---------------------------------------------------------------------------
template<int EPI>
__global__ void __launch_bounds__(256)
gemm_wmma(const _Float16* __restrict__ A, int lda,
          const _Float16* __restrict__ BT, int ldb,
          const float* __restrict__ bias,
          void* __restrict__ Cout, int ldc,
          int M, int N, int K,
          size_t sA0, size_t sA1, size_t sB0, size_t sB1,
          size_t sC0, size_t sC1, int zdiv)
{
    __shared__ _Float16 As[2][128 * 32];   // [row][k]  8 KB per buffer
    __shared__ _Float16 Bs[2][128 * 32];   // [col][k]  8 KB per buffer

    const int tid   = threadIdx.x;
    const int lane  = tid & 31;
    const int wave  = tid >> 5;
    const int waveM = wave >> 1;        // 0..3
    const int waveN = wave & 1;         // 0..1
    const int m16   = lane & 15;
    const int h2    = lane >> 4;        // lane half select

    const int rowBase = blockIdx.y * 128;
    const int colBase = blockIdx.x * 128;

    const int z  = blockIdx.z;
    const int zb = z / zdiv;
    const int zh = z - zb * zdiv;
    A  += (size_t)zb * sA0 + (size_t)zh * sA1;
    BT += (size_t)zb * sB0 + (size_t)zh * sB1;
    const size_t cOff = (size_t)zb * sC0 + (size_t)zh * sC1;

    // staging assignment: thread -> one 32B chunk of the A tile and B tile
    const int sr = tid >> 1;            // 0..127 (row of A tile / col of B tile)
    const int sc = (tid & 1) << 4;      // 0 or 16 halves

    auto stage = [&](int buf, int k0) {
        {   // A tile: rows are contiguous 64B runs of A
            const _Float16* gp = A + (size_t)(rowBase + sr) * lda + k0 + sc;
            _Float16*       lp = As[buf] + sr * 32 + sc;
            cp16_g2l(gp,     lp);
            cp16_g2l(gp + 8, lp + 8);
        }
        {   // B tile: rows are contiguous 64B runs of BT (already transposed)
            _Float16* lp = Bs[buf] + sr * 32 + sc;
            if (colBase + sr < N) {
                const _Float16* gp = BT + (size_t)(colBase + sr) * ldb + k0 + sc;
                cp16_g2l(gp,     lp);
                cp16_g2l(gp + 8, lp + 8);
            } else {
                const v4u zz = {0u, 0u, 0u, 0u};
                ((v4u*)lp)[0] = zz;
                ((v4u*)lp)[1] = zz;
            }
        }
    };

    v8f acc[2][4];
    #pragma unroll
    for (int mi = 0; mi < 2; ++mi)
        #pragma unroll
        for (int ni = 0; ni < 4; ++ni)
            acc[mi][ni] = v8f{0.f,0.f,0.f,0.f,0.f,0.f,0.f,0.f};

    stage(0, 0);                       // prologue prefetch

    const int steps = K / 32;
    for (int it = 0; it < steps; ++it) {
        const int cur = it & 1;
        wait_async_all();              // my async stores into buf[cur] done
        __syncthreads();               // everyone's buf[cur] ready; buf[cur^1] free
        if (it + 1 < steps) stage(cur ^ 1, (it + 1) * 32);   // prefetch next

        // ---- fragments per ISA layout, 8 WMMAs ----
        F16Frag af[2], bf[4];
        #pragma unroll
        for (int mi = 0; mi < 2; ++mi) {
            const _Float16* p = As[cur] + (waveM * 32 + mi * 16 + m16) * 32;
            af[mi].u[0] = *(const v4u*)(p + 8 * h2);        // K = 8h..8h+7
            af[mi].u[1] = *(const v4u*)(p + 16 + 8 * h2);   // K = 16+8h..
        }
        #pragma unroll
        for (int ni = 0; ni < 4; ++ni) {
            const _Float16* p = Bs[cur] + (waveN * 64 + ni * 16 + m16) * 32;
            bf[ni].u[0] = *(const v4u*)(p + 8 * h2);        // K = 8h..8h+7
            bf[ni].u[1] = *(const v4u*)(p + 16 + 8 * h2);   // K = 16+8h..
        }
        #pragma unroll
        for (int mi = 0; mi < 2; ++mi)
            #pragma unroll
            for (int ni = 0; ni < 4; ++ni)
                acc[mi][ni] = __builtin_amdgcn_wmma_f32_16x16x32_f16(
                    false, af[mi].h, false, bf[ni].h,
                    (short)0, acc[mi][ni], false, false);
    }

    // ---- epilogue: C layout = VGPR g -> row (g + 8*h2), lane&15 -> col ----
    float*    Cf = (float*)Cout;
    _Float16* Ch = (_Float16*)Cout;
    #pragma unroll
    for (int mi = 0; mi < 2; ++mi) {
        #pragma unroll
        for (int ni = 0; ni < 4; ++ni) {
            const int col = colBase + waveN * 64 + ni * 16 + m16;
            if (col >= N) continue;
            const float bb = bias ? bias[col] : 0.f;
            #pragma unroll
            for (int g = 0; g < 8; ++g) {
                const int row = rowBase + waveM * 32 + mi * 16 + 8 * h2 + g;
                float v = acc[mi][ni][g] + bb;
                if (EPI == 3) v = gelu_exact(v);
                if (EPI == 0) {
                    Cf[cOff + (size_t)row * ldc + col] = v;
                } else if (EPI == 4) {
                    // V^T store: row = b*SK + s  ->  out[(b*D + col)*SK + s]
                    const int b = row >> 10;           // /SK_
                    const int s = row & (SK_ - 1);
                    Ch[((size_t)b * D_ + col) * SK_ + s] = (_Float16)v;
                } else {
                    Ch[cOff + (size_t)row * ldc + col] = (_Float16)v;
                }
            }
        }
    }
}

// ---------------------------------------------------------------------------
// Attention scores:  S[b,h,q,k] = eff_scale * (qh . kh) (+ prev, in place)
// qh/kh are f16 [B, S, D] (head h at cols h*64..). HD=64 -> 2 K-steps.
// Fragments loaded straight from global: [seq, hd] layout matches the
// WMMA A/B fragment byte layout (two 16B loads per lane per fragment).
// ---------------------------------------------------------------------------
__global__ void __launch_bounds__(256)
attn_scores(const _Float16* __restrict__ qh, const _Float16* __restrict__ kh,
            float* __restrict__ scores,
            const float* __restrict__ scaleArr, const float* __restrict__ extraArr,
            int layer, int addPrev)
{
    const int z  = blockIdx.z;
    const int b  = z >> 3;          // /H_
    const int hh = z & 7;           // %H_

    float es = extraArr[layer];
    es = fminf(fmaxf(es, 0.01f), 50.f);
    const float eff = scaleArr[layer] * es;

    const int tid   = threadIdx.x;
    const int lane  = tid & 31;
    const int wave  = tid >> 5;
    const int waveM = wave >> 1;
    const int waveN = wave & 1;
    const int m16   = lane & 15;
    const int h2    = lane >> 4;

    const int m0 = blockIdx.y * 128 + waveM * 32;   // query rows
    const int n0 = blockIdx.x * 128 + waveN * 64;   // key cols

    v8f acc[2][4];
    #pragma unroll
    for (int mi = 0; mi < 2; ++mi)
        #pragma unroll
        for (int ni = 0; ni < 4; ++ni)
            acc[mi][ni] = v8f{0.f,0.f,0.f,0.f,0.f,0.f,0.f,0.f};

    #pragma unroll
    for (int ks = 0; ks < HD_; ks += 32) {
        F16Frag af[2], bf[4];
        #pragma unroll
        for (int mi = 0; mi < 2; ++mi) {
            const int qrow = m0 + mi * 16 + m16;
            const _Float16* p = qh + ((size_t)b * SQ_ + qrow) * D_ + hh * HD_ + ks;
            af[mi].u[0] = *(const v4u*)(p + 8 * h2);
            af[mi].u[1] = *(const v4u*)(p + 16 + 8 * h2);
        }
        #pragma unroll
        for (int ni = 0; ni < 4; ++ni) {
            const int key = n0 + ni * 16 + m16;
            const _Float16* p = kh + ((size_t)b * SK_ + key) * D_ + hh * HD_ + ks;
            bf[ni].u[0] = *(const v4u*)(p + 8 * h2);
            bf[ni].u[1] = *(const v4u*)(p + 16 + 8 * h2);
        }
        #pragma unroll
        for (int mi = 0; mi < 2; ++mi)
            #pragma unroll
            for (int ni = 0; ni < 4; ++ni)
                acc[mi][ni] = __builtin_amdgcn_wmma_f32_16x16x32_f16(
                    false, af[mi].h, false, bf[ni].h,
                    (short)0, acc[mi][ni], false, false);
    }

    #pragma unroll
    for (int mi = 0; mi < 2; ++mi) {
        #pragma unroll
        for (int ni = 0; ni < 4; ++ni) {
            const int col = n0 + ni * 16 + m16;
            #pragma unroll
            for (int g = 0; g < 8; ++g) {
                const int row = m0 + mi * 16 + 8 * h2 + g;
                const size_t idx = ((size_t)z * SQ_ + row) * SK_ + col;
                float v = acc[mi][ni][g] * eff;
                if (addPrev) v += scores[idx];
                scores[idx] = v;     // in place: scores buffer doubles as prev
            }
        }
    }
}

// ---------------------------------------------------------------------------
// Row softmax over SK_=1024; f16 probabilities out (pre-softmax scores kept).
// ---------------------------------------------------------------------------
__global__ void __launch_bounds__(128)
softmax_rows(const float* __restrict__ scores, _Float16* __restrict__ probs)
{
    __shared__ float red[128];
    const int tid = threadIdx.x;
    const size_t base = (size_t)blockIdx.x * SK_;

    float v[8];
    #pragma unroll
    for (int j = 0; j < 8; ++j) v[j] = scores[base + tid + j * 128];

    float mx = v[0];
    #pragma unroll
    for (int j = 1; j < 8; ++j) mx = fmaxf(mx, v[j]);
    red[tid] = mx; __syncthreads();
    for (int o = 64; o > 0; o >>= 1) {
        if (tid < o) red[tid] = fmaxf(red[tid], red[tid + o]);
        __syncthreads();
    }
    mx = red[0]; __syncthreads();

    float e[8], sum = 0.f;
    #pragma unroll
    for (int j = 0; j < 8; ++j) { e[j] = expf(v[j] - mx); sum += e[j]; }
    red[tid] = sum; __syncthreads();
    for (int o = 64; o > 0; o >>= 1) {
        if (tid < o) red[tid] += red[tid + o];
        __syncthreads();
    }
    const float inv = 1.f / red[0];
    #pragma unroll
    for (int j = 0; j < 8; ++j)
        probs[base + tid + j * 128] = (_Float16)(e[j] * inv);
}

// ---------------------------------------------------------------------------
// x = LayerNorm(x + softplus(gate) * y) * g + b   (row = D_ = 512)
// Also emits f16 copy of x for the next matrix-core GEMM.
// ---------------------------------------------------------------------------
__global__ void __launch_bounds__(128)
residual_ln(float* __restrict__ x, _Float16* __restrict__ xh,
            const float* __restrict__ y,
            const float* __restrict__ gateArr, int layer,
            const float* __restrict__ g, const float* __restrict__ bb)
{
    __shared__ float red[128];
    const int tid = threadIdx.x;
    const size_t base = (size_t)blockIdx.x * D_;

    const float gv = gateArr[layer];
    const float sp = (gv > 20.f) ? gv : log1pf(expf(gv));

    float xv[4];
    float sum = 0.f;
    #pragma unroll
    for (int j = 0; j < 4; ++j) {
        const int idx = tid + j * 128;
        xv[j] = x[base + idx] + sp * y[base + idx];
        sum += xv[j];
    }
    red[tid] = sum; __syncthreads();
    for (int o = 64; o > 0; o >>= 1) {
        if (tid < o) red[tid] += red[tid + o];
        __syncthreads();
    }
    const float mu = red[0] * (1.f / D_);
    __syncthreads();

    float vs = 0.f;
    #pragma unroll
    for (int j = 0; j < 4; ++j) { const float d = xv[j] - mu; vs += d * d; }
    red[tid] = vs; __syncthreads();
    for (int o = 64; o > 0; o >>= 1) {
        if (tid < o) red[tid] += red[tid + o];
        __syncthreads();
    }
    const float rs = rsqrtf(red[0] * (1.f / D_) + 1e-5f);

    #pragma unroll
    for (int j = 0; j < 4; ++j) {
        const int idx = tid + j * 128;
        const float o = (xv[j] - mu) * rs * g[idx] + bb[idx];
        x[base + idx]  = o;
        xh[base + idx] = (_Float16)o;
    }
}

// ---------------------------------------------------------------------------
// Weight transpose + f32->f16:  W[K][N] -> WT[N][K]   (K,N multiples of 32)
// ---------------------------------------------------------------------------
__global__ void __launch_bounds__(256)
transpose_w(const float* __restrict__ W, _Float16* __restrict__ WT, int K, int N)
{
    __shared__ float tile[32][33];
    const int tx = threadIdx.x & 31;
    const int ty = threadIdx.x >> 5;        // 0..7
    const int k0 = blockIdx.y * 32;
    const int n0 = blockIdx.x * 32;
    #pragma unroll
    for (int j = 0; j < 4; ++j)
        tile[ty + j * 8][tx] = W[(size_t)(k0 + ty + j * 8) * N + n0 + tx];
    __syncthreads();
    #pragma unroll
    for (int j = 0; j < 4; ++j)
        WT[(size_t)(n0 + ty + j * 8) * K + k0 + tx] =
            (_Float16)tile[tx][ty + j * 8];
}

// ---------------------------------------------------------------------------
// Flat f32 -> f16 convert
// ---------------------------------------------------------------------------
__global__ void __launch_bounds__(256)
convert_f16(const float* __restrict__ in, _Float16* __restrict__ out, size_t n)
{
    const size_t i = ((size_t)blockIdx.x * 256 + threadIdx.x) * 4;
    if (i + 3 < n) {
        #pragma unroll
        for (int j = 0; j < 4; ++j) out[i + j] = (_Float16)in[i + j];
    }
}

// ---------------------------------------------------------------------------
// Host orchestration
// ---------------------------------------------------------------------------
extern "C" void kernel_launch(void* const* d_in, const int* in_sizes, int n_in,
                              void* d_out, int out_size, void* d_ws, size_t ws_size,
                              hipStream_t stream)
{
    (void)in_sizes; (void)n_in; (void)out_size; (void)ws_size;

    const float* q_in  = (const float*)d_in[0];
    const float* k_in  = (const float*)d_in[1];
    const float* v_in  = (const float*)d_in[2];
    const float* WQ    = (const float*)d_in[3];
    const float* bQ    = (const float*)d_in[4];
    const float* WK    = (const float*)d_in[5];
    const float* bK    = (const float*)d_in[6];
    const float* WV    = (const float*)d_in[7];
    const float* bV    = (const float*)d_in[8];
    const float* WO    = (const float*)d_in[9];
    const float* bO    = (const float*)d_in[10];
    const float* Wf1   = (const float*)d_in[11];
    const float* bf1   = (const float*)d_in[12];
    const float* Wf2   = (const float*)d_in[13];
    const float* bf2   = (const float*)d_in[14];
    const float* ln1g  = (const float*)d_in[15];
    const float* ln1b  = (const float*)d_in[16];
    const float* ln2g  = (const float*)d_in[17];
    const float* ln2b  = (const float*)d_in[18];
    const float* scale = (const float*)d_in[19];
    const float* extra = (const float*)d_in[20];
    const float* gAttn = (const float*)d_in[21];
    const float* gFfn  = (const float*)d_in[22];

    const size_t xN  = (size_t)B_ * SQ_ * D_;          // 4M elements
    const size_t scN = (size_t)B_ * H_ * SQ_ * (size_t)SK_;
    const size_t ffN = (size_t)B_ * SQ_ * DFF_;

    auto align = [](size_t v) { return (v + 255) & ~(size_t)255; };
    char* w = (char*)d_ws;
    size_t off = 0;
    float*    x      = (float*)(w + off);    off = align(off + xN * 4);
    _Float16* xh     = (_Float16*)(w + off); off = align(off + xN * 2);
    _Float16* kinh   = (_Float16*)(w + off); off = align(off + xN * 2);
    _Float16* vinh   = (_Float16*)(w + off); off = align(off + xN * 2);
    _Float16* qh     = (_Float16*)(w + off); off = align(off + xN * 2);
    _Float16* kh     = (_Float16*)(w + off); off = align(off + xN * 2);
    _Float16* vhT    = (_Float16*)(w + off); off = align(off + xN * 2);   // [B][D][SK]
    _Float16* attnoh = (_Float16*)(w + off); off = align(off + xN * 2);
    float*    proj   = (float*)(w + off);    off = align(off + xN * 4);
    _Float16* ffh    = (_Float16*)(w + off); off = align(off + ffN * 2);
    float*    scores = (float*)(w + off);    off = align(off + scN * 4);
    _Float16* probs  = (_Float16*)(w + off); off = align(off + scN * 2);
    _Float16* WQT    = (_Float16*)(w + off); off = align(off + (size_t)D_ * D_ * 2);
    _Float16* WKT    = (_Float16*)(w + off); off = align(off + (size_t)D_ * D_ * 2);
    _Float16* WVT    = (_Float16*)(w + off); off = align(off + (size_t)D_ * D_ * 2);
    _Float16* WOT    = (_Float16*)(w + off); off = align(off + (size_t)D_ * D_ * 2);
    _Float16* W1T    = (_Float16*)(w + off); off = align(off + (size_t)D_ * DFF_ * 2);
    _Float16* W2T    = (_Float16*)(w + off); off = align(off + (size_t)D_ * DFF_ * 2);

    (void)hipMemcpyAsync(x, q_in, xN * 4, hipMemcpyDeviceToDevice, stream);

    const dim3 blk(256);
    const int Mrows = B_ * SQ_;
    const dim3 gCvt((unsigned)(xN / (256 * 4)));
    convert_f16<<<gCvt, blk, 0, stream>>>(q_in, xh, xN);
    convert_f16<<<gCvt, blk, 0, stream>>>(k_in, kinh, xN);
    convert_f16<<<gCvt, blk, 0, stream>>>(v_in, vinh, xN);

    for (int i = 0; i < L_; ++i) {
        // ---- weights -> transposed f16 ----
        transpose_w<<<dim3(D_/32,  D_/32),  blk, 0, stream>>>(WQ + (size_t)i*D_*D_,  WQT, D_,  D_);
        transpose_w<<<dim3(D_/32,  D_/32),  blk, 0, stream>>>(WK + (size_t)i*D_*D_,  WKT, D_,  D_);
        transpose_w<<<dim3(D_/32,  D_/32),  blk, 0, stream>>>(WV + (size_t)i*D_*D_,  WVT, D_,  D_);
        transpose_w<<<dim3(D_/32,  D_/32),  blk, 0, stream>>>(WO + (size_t)i*D_*D_,  WOT, D_,  D_);
        transpose_w<<<dim3(DFF_/32, D_/32), blk, 0, stream>>>(Wf1 + (size_t)i*D_*DFF_, W1T, D_,  DFF_);
        transpose_w<<<dim3(D_/32,  DFF_/32),blk, 0, stream>>>(Wf2 + (size_t)i*DFF_*D_, W2T, DFF_, D_);

        // ---- QKV projections (all matrix-core, async-staged) ----
        dim3 gP(D_ / 128, Mrows / 128, 1);
        gemm_wmma<2><<<gP, blk, 0, stream>>>(
            xh, D_, WQT, D_, bQ + (size_t)i * D_, qh, D_,
            Mrows, D_, D_, 0, 0, 0, 0, 0, 0, 1);
        gemm_wmma<2><<<gP, blk, 0, stream>>>(
            kinh, D_, WKT, D_, bK + (size_t)i * D_, kh, D_,
            Mrows, D_, D_, 0, 0, 0, 0, 0, 0, 1);
        gemm_wmma<4><<<gP, blk, 0, stream>>>(           // writes V^T directly
            vinh, D_, WVT, D_, bV + (size_t)i * D_, vhT, SK_,
            Mrows, D_, D_, 0, 0, 0, 0, 0, 0, 1);

        // ---- scores = eff_scale * Q.K^T (+ prev, in place) ----
        dim3 gS(SK_ / 128, SQ_ / 128, B_ * H_);
        attn_scores<<<gS, blk, 0, stream>>>(qh, kh, scores, scale, extra,
                                            i, (i > 0) ? 1 : 0);

        // ---- softmax -> f16 probabilities ----
        softmax_rows<<<dim3(B_ * H_ * SQ_), dim3(128), 0, stream>>>(scores, probs);

        // ---- attn_out = probs @ V   (batched z = b*H + h; B = V^T rows) ----
        dim3 gPV(1, SQ_ / 128, B_ * H_);
        gemm_wmma<2><<<gPV, blk, 0, stream>>>(
            probs, SK_, vhT, SK_, nullptr, attnoh, D_,
            SQ_, HD_, SK_,
            (size_t)H_ * SQ_ * SK_, (size_t)SQ_ * SK_,   // A strides (b,h)
            (size_t)D_ * SK_, (size_t)HD_ * SK_,         // B strides (b,h)
            (size_t)SQ_ * D_, (size_t)HD_,               // C strides (b,h)
            H_);

        // ---- output projection (f32 out for residual) ----
        gemm_wmma<0><<<gP, blk, 0, stream>>>(
            attnoh, D_, WOT, D_, bO + (size_t)i * D_, proj, D_,
            Mrows, D_, D_, 0, 0, 0, 0, 0, 0, 1);

        // ---- gated residual + LN1 (also refreshes xh) ----
        residual_ln<<<dim3(Mrows), dim3(128), 0, stream>>>(
            x, xh, proj, gAttn, i, ln1g + (size_t)i * D_, ln1b + (size_t)i * D_);

        // ---- FFN1 (gelu epilogue, f16 out) ----
        dim3 gF1(DFF_ / 128, Mrows / 128, 1);
        gemm_wmma<3><<<gF1, blk, 0, stream>>>(
            xh, D_, W1T, D_, bf1 + (size_t)i * DFF_, ffh, DFF_,
            Mrows, DFF_, D_, 0, 0, 0, 0, 0, 0, 1);

        // ---- FFN2 (f32 out) ----
        dim3 gF2(D_ / 128, Mrows / 128, 1);
        gemm_wmma<0><<<gF2, blk, 0, stream>>>(
            ffh, DFF_, W2T, DFF_, bf2 + (size_t)i * D_, proj, D_,
            Mrows, D_, DFF_, 0, 0, 0, 0, 0, 0, 1);

        // ---- gated residual + LN2 ----
        residual_ln<<<dim3(Mrows), dim3(128), 0, stream>>>(
            x, xh, proj, gFfn, i, ln2g + (size_t)i * D_, ln2b + (size_t)i * D_);
    }

    (void)hipMemcpyAsync(d_out, x, xN * 4, hipMemcpyDeviceToDevice, stream);
}
